// AttSRU_65738769433394
// MI455X (gfx1250) — matrix operand order repaired and checked by reading, not compiled, not run on hardware
//
#include <hip/hip_runtime.h>
#include <hip/hip_bf16.h>
#include <stdint.h>

// ---------------------------------------------------------------------------
// AttSRU fused pipeline for gfx1250 (MI455X), wave32, WMMA bf16.
// ---------------------------------------------------------------------------

typedef __bf16 bf16;
typedef __attribute__((ext_vector_type(16))) bf16  v16bf;
typedef __attribute__((ext_vector_type(8)))  bf16  v8bf;
typedef __attribute__((ext_vector_type(8)))  float v8f;
typedef __attribute__((ext_vector_type(4)))  float v4f;

#define T_   128
#define B_   32
#define S_   100
#define D_   256
#define M_   4096            // T*B
#define EPS_ 1e-6f

__device__ __forceinline__ bf16  f2b(float f) { return (bf16)f; }
__device__ __forceinline__ float sigm(float x){ return 1.0f / (1.0f + __expf(-x)); }

__device__ __forceinline__ float wave_sum(float x){
  #pragma unroll
  for (int off = 16; off > 0; off >>= 1) x += __shfl_xor(x, off, 32);
  return x;
}
__device__ __forceinline__ float wave_max(float x){
  #pragma unroll
  for (int off = 16; off > 0; off >>= 1) x = fmaxf(x, __shfl_xor(x, off, 32));
  return x;
}

// ---------------------------------------------------------------------------
// Weight prep: W[K,N] f32 (row-major) -> Wt[N,K] bf16 (transposed), so a WMMA
// B-fragment (one N column, 16 consecutive K) is a contiguous 32-byte load.
// ---------------------------------------------------------------------------
__global__ void wprep(const float* __restrict__ W, bf16* __restrict__ Wt, int K, int N){
  int idx = blockIdx.x * blockDim.x + threadIdx.x;
  if (idx >= K * N) return;
  int k = idx / N, n = idx - k * N;
  Wt[(size_t)n * K + k] = f2b(W[idx]);
}

// ---------------------------------------------------------------------------
// Fused Y = LN_N(A @ W) (optionally sigmoid on cols < 512 when MODE==1).
// A: [M,256] f32.  Wt: [N,256] bf16 transposed.  One WG = 16 rows x N cols.
// 8 waves; tiles-per-wave is a compile-time constant so all loops around the
// WMMA body are uniform (no exec-masked loops around v_wmma).
// ---------------------------------------------------------------------------
template<int N, int MODE>
__global__ void __launch_bounds__(256)
gemm_ln(const float* __restrict__ A, const bf16* __restrict__ Wt,
        const float* __restrict__ g, const float* __restrict__ bia,
        float* __restrict__ Y){
  constexpr int NT  = N / 16;   // 16-wide output tiles
  constexpr int TPW = NT / 8;   // tiles per wave (2 for N=256, 6 for N=768)
  extern __shared__ char smem[];
  bf16*  As = (bf16*)smem;                       // 16*256 bf16 = 8 KB
  float* Ys = (float*)(smem + 16 * 256 * 2);     // 16*N f32
  const int tid = threadIdx.x, wid = tid >> 5, ln = tid & 31;
  const int m0 = blockIdx.x * 16;

  // A tile -> bf16 LDS
  #pragma unroll
  for (int i0 = 0; i0 < 16; ++i0){
    int i = i0 * 256 + tid;
    As[i] = f2b(A[(size_t)(m0 + (i >> 8)) * 256 + (i & 255)]);
  }
  __syncthreads();

  // Per-lane A fragment addressing (contiguous k order, same as B):
  // lanes 0-15: M=lane, halves 0..15 -> K0..15
  // lanes 16-31: same with K offset +16
  const int am = (ln & 15);
  const int ak = (ln >> 4) * 16;
  v16bf afr[8];
  #pragma unroll
  for (int k = 0; k < 8; ++k){
    const int k0 = k * 32;
    v8bf lo = *(const v8bf*)&As[am * 256 + k0 + ak];
    v8bf hi = *(const v8bf*)&As[am * 256 + k0 + 8 + ak];
    v16bf a;
    #pragma unroll
    for (int i = 0; i < 8; ++i){ a[i] = lo[i]; a[8 + i] = hi[i]; }
    afr[k] = a;
  }

  const int kb = (ln >> 4) * 16;
  #pragma unroll
  for (int t = 0; t < TPW; ++t){
    const int nt = wid + t * 8;                  // wave-uniform
    const int nn = nt * 16 + (ln & 15);
    v8f acc = {};
    #pragma unroll
    for (int k = 0; k < 8; ++k){
      v16bf b = *(const v16bf*)&Wt[(size_t)nn * 256 + k * 32 + kb];
      acc = __builtin_amdgcn_wmma_f32_16x16x32_bf16(false, afr[k], false, b,
                                                    (short)0, acc, false, false);
    }
    // C layout: VGPR j -> row j (lanes 0-15) / row j+8 (lanes 16-31); col = lane&15
    const int crow = 8 * (ln >> 4);
    const int ccol = nt * 16 + (ln & 15);
    #pragma unroll
    for (int j = 0; j < 8; ++j) Ys[(crow + j) * N + ccol] = acc[j];
  }
  __syncthreads();

  // LayerNorm per row over N; 2 rows per wave, uniform trip count.
  #pragma unroll
  for (int rr = 0; rr < 2; ++rr){
    const int r = wid * 2 + rr;
    float s = 0.f, sq = 0.f;
    #pragma unroll
    for (int c0 = 0; c0 < N / 32; ++c0){
      float v = Ys[r * N + c0 * 32 + ln]; s += v; sq += v * v;
    }
    s = wave_sum(s); sq = wave_sum(sq);
    float mean = s * (1.f / (float)N);
    float rstd = rsqrtf(sq * (1.f / (float)N) - mean * mean + EPS_);
    #pragma unroll
    for (int c0 = 0; c0 < N / 32; ++c0){
      const int c = c0 * 32 + ln;
      float v = (Ys[r * N + c] - mean) * rstd * g[c] + bia[c];
      if (MODE == 1 && c < 512) v = sigm(v);
      Y[(size_t)(m0 + r) * N + c] = v;
    }
  }
}

// ---------------------------------------------------------------------------
// SRU elementwise scan over T (sequential), parallel over B*D = 8192 lanes.
// preact layout [T*B, 768]: [0,256)=sigmoid(z), [256,512)=sigmoid(hgate),
// [512,768)=x_t.
// ---------------------------------------------------------------------------
__global__ void sru_scan(const float* __restrict__ preact, const float* __restrict__ h0,
                         float* __restrict__ ss, float* __restrict__ final_h){
  int t = blockIdx.x * blockDim.x + threadIdx.x;
  if (t >= B_ * D_) return;
  int bb = t >> 8, d = t & 255;
  float h = h0[t];
  for (int tt = 0; tt < T_; ++tt){
    size_t row = (size_t)tt * B_ + bb;
    float z = preact[row * 768 + d];
    float x = preact[row * 768 + 512 + d];
    h = (1.f - z) * h + z * x;
    ss[row * 256 + d] = h;
  }
  final_h[t] = h;
}

// ---------------------------------------------------------------------------
// Fused per-n attention block (one WG per n, 4096 WGs):
//   enc[n] (100x256 f32) --async--> LDS; pctx = LN(enc @ Wenc) kept in LDS
//   (bf16); scores = v . tanh(q + pctx); softmax; attn_out = attn @ pctx / 16.
// pctx never touches HBM: saves ~1.3 GB of round-trips.
// ---------------------------------------------------------------------------
__global__ void __launch_bounds__(256, 1)
attn_fused(const float* __restrict__ enc,    // [M, S, D] f32
           const bf16*  __restrict__ WencT,  // [256,256] bf16 transposed
           const float* __restrict__ q,      // [M,256] f32
           const float* __restrict__ vatt,   // [256]
           const float* __restrict__ g_enc, const float* __restrict__ b_enc,
           float* __restrict__ attn_out,     // [M,256]
           float* __restrict__ attn_probs){  // [M,100]
  extern __shared__ char smem[];
  float* encF = (float*)smem;                               // 112*256 f32 (rows 100..111 zero)
  bf16*  pctx = (bf16*)(smem + 112 * 256 * 4);              // 112*256 bf16
  float* red  = (float*)(smem + 112 * 256 * 4 + 112 * 256 * 2); // [0,128)=scores [128,256)=attn
  const int tid = threadIdx.x, wid = tid >> 5, ln = tid & 31;
  const int n = blockIdx.x;
  const uint64_t gbase = (uint64_t)(uintptr_t)(enc + (size_t)n * S_ * D_);

  // Async global->LDS copy of enc[n]: 100*256 f32 = 6400 x 16B chunks.
  #pragma unroll
  for (int c = 0; c < 25; ++c){
    const int i = c * 256 + tid;
    uint32_t ldsoff = (uint32_t)(uintptr_t)(void*)&encF[i * 4];
    uint32_t voff   = (uint32_t)(i * 16);
    asm volatile("global_load_async_to_lds_b128 %0, %1, %2"
                 :: "v"(ldsoff), "v"(voff), "s"(gbase) : "memory");
  }
  // Zero-pad rows 100..111 so M-tile 6 (rows 96..111) is clean.
  #pragma unroll
  for (int c = 0; c < 12; ++c) encF[100 * 256 + c * 256 + tid] = 0.f;
  asm volatile("s_wait_asynccnt 0x0" ::: "memory");
  __syncthreads();

  // GEMM: pctx[112x256] = encF @ Wenc. 7 M-tiles; each wave owns 2 N-tiles.
  const int am = (ln & 15);
  const int ak = (ln >> 4) * 16;
  const int kb = (ln >> 4) * 16;
  const int nn0 = wid * 16 + (ln & 15);
  const int nn1 = (wid + 8) * 16 + (ln & 15);

  #pragma unroll
  for (int mt = 0; mt < 7; ++mt){
    v16bf afr[8];
    #pragma unroll
    for (int k = 0; k < 8; ++k){
      const float* ap = &encF[(mt * 16 + am) * 256 + k * 32 + ak];
      v4f f0 = *(const v4f*)(ap);
      v4f f1 = *(const v4f*)(ap + 4);
      v4f f2 = *(const v4f*)(ap + 8);
      v4f f3 = *(const v4f*)(ap + 12);
      v16bf a;
      #pragma unroll
      for (int i = 0; i < 4; ++i){
        a[i]      = f2b(f0[i]);
        a[4 + i]  = f2b(f1[i]);
        a[8 + i]  = f2b(f2[i]);
        a[12 + i] = f2b(f3[i]);
      }
      afr[k] = a;
    }
    v8f acc0 = {}, acc1 = {};
    #pragma unroll
    for (int k = 0; k < 8; ++k){
      v16bf b0 = *(const v16bf*)&WencT[(size_t)nn0 * 256 + k * 32 + kb];
      v16bf b1 = *(const v16bf*)&WencT[(size_t)nn1 * 256 + k * 32 + kb];
      acc0 = __builtin_amdgcn_wmma_f32_16x16x32_bf16(false, afr[k], false, b0,
                                                     (short)0, acc0, false, false);
      acc1 = __builtin_amdgcn_wmma_f32_16x16x32_bf16(false, afr[k], false, b1,
                                                     (short)0, acc1, false, false);
    }
    const int crow = mt * 16 + 8 * (ln >> 4);
    #pragma unroll
    for (int j = 0; j < 8; ++j){
      pctx[(crow + j) * 256 + wid * 16 + (ln & 15)]       = f2b(acc0[j]);
      pctx[(crow + j) * 256 + (wid + 8) * 16 + (ln & 15)] = f2b(acc1[j]);
    }
  }
  __syncthreads();

  // LayerNorm pctx rows 0..99 in place (bf16), one row per wave round-robin.
  for (int r = wid; r < S_; r += 8){
    float s = 0.f, sq = 0.f;
    #pragma unroll
    for (int c0 = 0; c0 < 8; ++c0){
      float v = (float)pctx[r * 256 + c0 * 32 + ln]; s += v; sq += v * v;
    }
    s = wave_sum(s); sq = wave_sum(sq);
    float mean = s * (1.f / 256.f);
    float rstd = rsqrtf(sq * (1.f / 256.f) - mean * mean + EPS_);
    #pragma unroll
    for (int c0 = 0; c0 < 8; ++c0){
      const int c = c0 * 32 + ln;
      float v = ((float)pctx[r * 256 + c] - mean) * rstd * g_enc[c] + b_enc[c];
      pctx[r * 256 + c] = f2b(v);
    }
  }
  __syncthreads();

  // Scores: each wave handles rows s = wid, wid+8, ...; lane covers 8 d's.
  float qv[8], vv[8];
  #pragma unroll
  for (int i = 0; i < 8; ++i){
    qv[i] = q[(size_t)n * 256 + ln * 8 + i];
    vv[i] = vatt[ln * 8 + i];
  }
  for (int s = wid; s < S_; s += 8){
    v8bf p = *(const v8bf*)&pctx[s * 256 + ln * 8];
    float sc = 0.f;
    #pragma unroll
    for (int i = 0; i < 8; ++i) sc += vv[i] * tanhf(qv[i] + (float)p[i]);
    sc = wave_sum(sc);
    if (ln == 0) red[s] = sc;
  }
  __syncthreads();

  // Softmax over S=100 by wave 0.
  if (wid == 0){
    float vals[4], mx = -3.0e38f;
    #pragma unroll
    for (int i = 0; i < 4; ++i){
      int s = ln + i * 32;
      vals[i] = (s < S_) ? red[s] : -3.0e38f;
      mx = fmaxf(mx, vals[i]);
    }
    mx = wave_max(mx);
    float sum = 0.f;
    #pragma unroll
    for (int i = 0; i < 4; ++i){
      vals[i] = (ln + i * 32 < S_) ? __expf(vals[i] - mx) : 0.f;
      sum += vals[i];
    }
    sum = wave_sum(sum);
    float inv = 1.f / sum;
    #pragma unroll
    for (int i = 0; i < 4; ++i){
      int s = ln + i * 32;
      if (s < S_){
        float a = vals[i] * inv;
        red[128 + s] = a;
        attn_probs[(size_t)n * S_ + s] = a;
      }
    }
  }
  __syncthreads();

  // attn_out[n,d] = (1/sqrt(256)) * sum_s attn[s] * pctx[s,d]; thread = d.
  {
    const int d = tid;
    float acc = 0.f;
    for (int s = 0; s < S_; ++s) acc += red[128 + s] * (float)pctx[s * 256 + d];
    attn_out[(size_t)n * 256 + d] = acc * 0.0625f;
  }
}

// ---------------------------------------------------------------------------
// out = (1-hgate)*tanh(yh+yc) + hgate*prev
// ---------------------------------------------------------------------------
__global__ void final_out(const float* __restrict__ yh, const float* __restrict__ yc,
                          const float* __restrict__ preact, const float* __restrict__ prev,
                          float* __restrict__ out){
  size_t i = (size_t)blockIdx.x * blockDim.x + threadIdx.x;
  if (i >= (size_t)M_ * 256) return;
  size_t row = i >> 8; int d = (int)(i & 255);
  float hg = preact[row * 768 + 256 + d];
  float o = tanhf(yh[i] + yc[i]);
  out[i] = (1.f - hg) * o + hg * prev[i];
}

// ---------------------------------------------------------------------------
extern "C" void kernel_launch(void* const* d_in, const int* in_sizes, int n_in,
                              void* d_out, int out_size, void* d_ws, size_t ws_size,
                              hipStream_t stream){
  (void)in_sizes; (void)n_in; (void)out_size; (void)ws_size;
  const float* prev   = (const float*)d_in[0];
  const float* hidden = (const float*)d_in[1];
  const float* enc    = (const float*)d_in[2];
  const float* W_in   = (const float*)d_in[3];
  const float* W_hid  = (const float*)d_in[4];
  const float* W_ctx  = (const float*)d_in[5];
  const float* W_enc  = (const float*)d_in[6];
  const float* W_q    = (const float*)d_in[7];
  const float* v_att  = (const float*)d_in[8];
  const float* g_pre  = (const float*)d_in[9];
  const float* b_pre  = (const float*)d_in[10];
  const float* g_enc  = (const float*)d_in[11];
  const float* b_enc  = (const float*)d_in[12];
  const float* g_h    = (const float*)d_in[13];
  const float* b_h    = (const float*)d_in[14];
  const float* g_c    = (const float*)d_in[15];
  const float* b_c    = (const float*)d_in[16];
  const float* g_q    = (const float*)d_in[17];
  const float* b_q    = (const float*)d_in[18];

  char* ws = (char*)d_ws;
  bf16*  WinT   = (bf16*)(ws + 0);          // 768*256*2 = 393216
  bf16*  WencT  = (bf16*)(ws + 393216);     // 131072
  bf16*  WqT    = (bf16*)(ws + 524288);     // 131072
  bf16*  WhidT  = (bf16*)(ws + 655360);     // 131072
  bf16*  WctxT  = (bf16*)(ws + 786432);     // 131072
  float* preact = (float*)(ws + 917504);    // 4096*768*4 = 12582912
  float* ss     = (float*)(ws + 13500416);  // 4096*256*4 = 4194304
  float* qbuf   = (float*)(ws + 17694720);  // 4194304
  float* aout   = (float*)(ws + 21889024);  // 4194304
  float* yh     = (float*)(ws + 26083328);  // 4194304
  float* yc     = (float*)(ws + 30277632);  // 4194304  (total ~34.5 MB)

  float* out   = (float*)d_out;
  float* fhout = out + (size_t)M_ * 256;    // final_h [B,D]
  float* attnp = fhout + (size_t)B_ * 256;  // attn [T*B,S]

  // Weight transpose + bf16 convert.
  wprep<<<(256 * 768 + 255) / 256, 256, 0, stream>>>(W_in,  WinT,  256, 768);
  wprep<<<(256 * 256 + 255) / 256, 256, 0, stream>>>(W_enc, WencT, 256, 256);
  wprep<<<(256 * 256 + 255) / 256, 256, 0, stream>>>(W_q,   WqT,   256, 256);
  wprep<<<(256 * 256 + 255) / 256, 256, 0, stream>>>(W_hid, WhidT, 256, 256);
  wprep<<<(256 * 256 + 255) / 256, 256, 0, stream>>>(W_ctx, WctxT, 256, 256);

  const size_t sm_g768 = 16 * 256 * 2 + 16 * 768 * 4;  // 57344
  const size_t sm_g256 = 16 * 256 * 2 + 16 * 256 * 4;  // 24576

  // preact = LN(prev @ W_in), sigmoid on z/hgate thirds.
  gemm_ln<768, 1><<<M_ / 16, 256, sm_g768, stream>>>(prev, WinT, g_pre, b_pre, preact);
  // SRU scan.
  sru_scan<<<(B_ * D_) / 256, 256, 0, stream>>>(preact, hidden, ss, fhout);
  // q = LN(ss @ W_q).
  gemm_ln<256, 0><<<M_ / 16, 256, sm_g256, stream>>>(ss, WqT, g_q, b_q, qbuf);

  // Fused encoder projection + MLP attention (pctx lives in LDS only).
  const size_t sm_attn = 112 * 256 * 4 + 112 * 256 * 2 + 256 * 4;  // 173056
  hipFuncSetAttribute((const void*)attn_fused,
                      hipFuncAttributeMaxDynamicSharedMemorySize, (int)sm_attn);
  attn_fused<<<M_, 256, sm_attn, stream>>>(enc, WencT, qbuf, v_att, g_enc, b_enc,
                                           aout, attnp);

  // Output projections + final gate.
  gemm_ln<256, 0><<<M_ / 16, 256, sm_g256, stream>>>(ss,   WhidT, g_h, b_h, yh);
  gemm_ln<256, 0><<<M_ / 16, 256, sm_g256, stream>>>(aout, WctxT, g_c, b_c, yc);
  final_out<<<(M_ * 256) / 256, 256, 0, stream>>>(yh, yc, preact, prev, out);
}
